// SparseGPT_25675314495502
// MI455X (gfx1250) — matrix operand order repaired and checked by reading, not compiled, not run on hardware
//
#include <hip/hip_runtime.h>
#include <hip/hip_bf16.h>
#include <math.h>

// Problem constants (match reference)
#define T_SEQ   2048
#define DMODEL  1024
#define NHEAD   16
#define DHEAD   64
#define DMLP    4096
#define NVOCAB  32000
#define NLAYER  4

typedef __attribute__((ext_vector_type(16))) __bf16          v16bf;
typedef __attribute__((ext_vector_type(8)))  float           v8f;
typedef __attribute__((ext_vector_type(16))) unsigned short  v16u;
typedef __attribute__((ext_vector_type(8)))  unsigned short  v8u;
typedef __attribute__((ext_vector_type(4)))  unsigned int    u32x4;
typedef __attribute__((ext_vector_type(8)))  int             i32x8;
typedef __attribute__((ext_vector_type(4)))  int             i32x4;

// Hardware bf16 convert (v_cvt_pk_bf16_f32), RNE — replaces 3-op integer RNE.
__device__ inline unsigned short f2bf(float f) {
  __bf16 h = (__bf16)f;
  return __builtin_bit_cast(unsigned short, h);
}

__device__ inline v8f wmma_bf16(v16u a, v16u b, v8f c) {
  // D = A(16x32 bf16) * B(32x16 bf16) + C(16x16 f32)
  return __builtin_amdgcn_wmma_f32_16x16x32_bf16(
      false, __builtin_bit_cast(v16bf, a),
      false, __builtin_bit_cast(v16bf, b),
      (short)0, c, false, false);
}

// ---------------------------------------------------------------------------
// x[t,:] = wte[idx[t],:] + wpe[t,:]
__global__ __launch_bounds__(256) void embed_kernel(
    const int* __restrict__ idx, const float* __restrict__ wte,
    const float* __restrict__ wpe, float* __restrict__ x) {
  int t = blockIdx.x;
  int tok = idx[t];
  for (int d = threadIdx.x; d < DMODEL; d += 256)
    x[(size_t)t * DMODEL + d] =
        wte[(size_t)tok * DMODEL + d] + wpe[(size_t)t * DMODEL + d];
}

// ---------------------------------------------------------------------------
// out[row,:] = bf16( x * rsqrt(mean(x^2)+eps) * w )
__global__ __launch_bounds__(256) void rmsnorm_kernel(
    const float* __restrict__ x, const float* __restrict__ w,
    unsigned short* __restrict__ out, int D) {
  __shared__ float red[256];
  int row = blockIdx.x, t = threadIdx.x;
  const float* xr = x + (size_t)row * D;
  float ss = 0.f;
  for (int i = t; i < D; i += 256) { float v = xr[i]; ss += v * v; }
  red[t] = ss; __syncthreads();
  for (int s = 128; s > 0; s >>= 1) {
    if (t < s) red[t] += red[t + s];
    __syncthreads();
  }
  float rs = rsqrtf(red[0] / (float)D + 1e-6f);
  for (int i = t; i < D; i += 256)
    out[(size_t)row * D + i] = f2bf(xr[i] * rs * w[i]);
}

// ---------------------------------------------------------------------------
// C[MxN] = act( A[MxK bf16] @ B[KxN f32] (+ R) )
// MODE 0: plain f32   MODE 1: +residual f32   MODE 2: exact gelu f32
// MODE 3: bf16 output (for QKV -> attention)
// Tile: 128x128, 8 waves, K-step 32, bf16 WMMA 16x16x32.
// A tile staged to LDS by the Tensor Data Mover (tensor_load_to_lds).
template <int MODE>
__global__ __launch_bounds__(256) void gemm_kernel(
    const unsigned short* __restrict__ A, const float* __restrict__ B,
    void* Cv, const float* R, int M, int N, int K) {
  __shared__ unsigned short lA[128 * 32];  // [row][k]
  __shared__ unsigned short lB[128 * 32];  // [col][k]  (pre-swizzled for B frags)
  const int t = threadIdx.x;
  const int lane = t & 31, wave = t >> 5;
  const int m0 = blockIdx.y * 128, n0 = blockIdx.x * 128;
  const int l16 = lane & 15, half = lane >> 4;

  v8f acc[8] = {};

  const int bcol = t & 127, bkh = (t >> 7) * 16;  // B stage: 1 col, 16 k's

  for (int k0 = 0; k0 < K; k0 += 32) {
    // ---- Stage A tile via TDM: 2D tile 32(elems) x 128(rows), 2B/elem ----
    if (wave == 0) {
      unsigned long long ga =
          (unsigned long long)(uintptr_t)(A + (size_t)m0 * K + k0);
      u32x4 g0;
      g0[0] = 1u;                                   // count=1 (valid, user mode)
      g0[1] = (unsigned)(uintptr_t)&lA[0];          // lds_addr (bytes)
      g0[2] = (unsigned)(ga & 0xFFFFFFFFu);         // global_addr[31:0]
      g0[3] = (unsigned)((ga >> 32) & 0x1FFFFFFu) | (2u << 30);  // addr[56:32]|type=2
      i32x8 g1;
      g1[0] = 0x00010000;                           // wg_mask=0, data_size=2B
      g1[1] = (int)((unsigned)(K & 0xFFFF) << 16);  // tensor_dim0[15:0]
      g1[2] = (int)(((unsigned)K >> 16) | ((unsigned)(M & 0xFFFF) << 16));
      g1[3] = (int)(((unsigned)M >> 16) | (32u << 16));  // tile_dim0 = 32
      g1[4] = 128;                                  // tile_dim1=128, tile_dim2=0
      g1[5] = K;                                    // tensor_dim0_stride[31:0]
      g1[6] = 0;                                    // stride hi / dim1_stride lo
      g1[7] = 0;
      i32x4 gz4 = {0, 0, 0, 0};
      i32x8 gz8 = {0, 0, 0, 0, 0, 0, 0, 0};
      __builtin_amdgcn_tensor_load_to_lds(g0, g1, gz4, gz4, gz8, 0);
      __builtin_amdgcn_s_wait_tensorcnt(0);
    }
    // ---- Stage B tile: coalesced f32 loads, hw convert, swizzle [col][k] ----
    v16u bv;
#pragma unroll
    for (int i = 0; i < 16; i++)
      bv[i] = f2bf(B[(size_t)(k0 + bkh + i) * N + n0 + bcol]);
    *(v16u*)&lB[bcol * 32 + bkh] = bv;
    __syncthreads();

    // A fragment: lane<16 -> row, K 0..15 ; lane>=16 -> K 16..31
    int ar = wave * 16 + l16;
    int ak = half * 16;
    v8u alo = *(const v8u*)&lA[ar * 32 + ak];
    v8u ahi = *(const v8u*)&lA[ar * 32 + 8 + ak];
    v16u af;
#pragma unroll
    for (int i = 0; i < 8; i++) { af[i] = alo[i]; af[i + 8] = ahi[i]; }

#pragma unroll
    for (int nt = 0; nt < 8; nt++) {
      // B fragment: lane = col, half-wave selects K half -> contiguous 32B in lB
      int col = nt * 16 + l16;
      v16u bf = *(const v16u*)&lB[col * 32 + half * 16];
      acc[nt] = wmma_bf16(af, bf, acc[nt]);
    }
    __syncthreads();
  }

  // C layout: VGPR j -> row j (+8 for upper half-wave); lane -> col
  int rbase = m0 + wave * 16 + half * 8;
  int cbase = n0 + l16;
#pragma unroll
  for (int nt = 0; nt < 8; nt++) {
#pragma unroll
    for (int j = 0; j < 8; j++) {
      size_t off = (size_t)(rbase + j) * N + cbase + nt * 16;
      float v = acc[nt][j];
      if (MODE == 3) {
        ((unsigned short*)Cv)[off] = f2bf(v);
      } else {
        if (MODE == 1) v += R[off];
        if (MODE == 2) v = 0.5f * v * (1.0f + erff(v * 0.70710678118f));
        ((float*)Cv)[off] = v;
      }
    }
  }
}

// ---------------------------------------------------------------------------
// Flash attention with sink logit. One wave per (16-query tile, head).
// qkv (bf16) layout: [T][3072] = [q(1024) | k(1024) | v(1024)], head h at h*64.
__global__ __launch_bounds__(32) void attn_kernel(
    const unsigned short* __restrict__ qkv, const float* __restrict__ sinkL,
    unsigned short* __restrict__ y) {
  __shared__ unsigned short lP[16 * 32];
  const int lane = threadIdx.x;
  const int qt = blockIdx.x, h = blockIdx.y;
  const int l16 = lane & 15, half = lane >> 4;
  const int qrow = qt * 16 + l16;
  const int k0 = half * 16;

  // Q as two A fragments (head-dim 0..31 and 32..63): pure 16-bit loads
  const unsigned short* qbase = qkv + (size_t)qrow * 3072 + h * 64;
  v16u qf[2];
#pragma unroll
  for (int f = 0; f < 2; f++) {
    v8u lo = *(const v8u*)(qbase + f * 32 + k0);
    v8u hi = *(const v8u*)(qbase + f * 32 + 8 + k0);
#pragma unroll
    for (int i = 0; i < 8; i++) { qf[f][i] = lo[i]; qf[f][i + 8] = hi[i]; }
  }

  float sink = sinkL[h];
  float m[8], l[8];
  v8f acc[4] = {};
#pragma unroll
  for (int j = 0; j < 8; j++) { m[j] = sink; l[j] = 1.0f; }  // sink column

  const int kmax = qt * 16 + 15;
  for (int kb = 0; kb * 32 <= kmax; kb++) {
    // ---- S = Q K^T for 32 keys (two 16-wide WMMA N-tiles) ----
    v8f s[2];
#pragma unroll
    for (int nt = 0; nt < 2; nt++) {
      int key = kb * 32 + nt * 16 + l16;
      const unsigned short* kbase =
          qkv + (size_t)key * 3072 + 1024 + h * 64 + half * 16;
      v16u kf0 = *(const v16u*)kbase;        // dims half*16 + 0..15
      v16u kf1 = *(const v16u*)(kbase + 32); // dims 32 + half*16 + 0..15
      v8f sv = {};
      sv = wmma_bf16(qf[0], kf0, sv);
      sv = wmma_bf16(qf[1], kf1, sv);
      s[nt] = sv;
    }
    // ---- online softmax (half-wave shuffle reductions per row) ----
    float p[2][8];
#pragma unroll
    for (int j = 0; j < 8; j++) {
      int row = qt * 16 + j + half * 8;
#pragma unroll
      for (int nt = 0; nt < 2; nt++) {
        int key = kb * 32 + nt * 16 + l16;
        float sv = s[nt][j] * 0.125f;   // 1/sqrt(64)
        if (key > row) sv = -1e30f;     // causal mask
        s[nt][j] = sv;
      }
      float mx = fmaxf(s[0][j], s[1][j]);
#pragma unroll
      for (int off = 8; off >= 1; off >>= 1)
        mx = fmaxf(mx, __shfl_xor(mx, off, 16));
      float mnew = fmaxf(m[j], mx);
      float sc = __expf(m[j] - mnew);
      float p0 = __expf(s[0][j] - mnew);
      float p1 = __expf(s[1][j] - mnew);
      float rs = p0 + p1;
#pragma unroll
      for (int off = 8; off >= 1; off >>= 1)
        rs += __shfl_xor(rs, off, 16);
      l[j] = l[j] * sc + rs;
      m[j] = mnew;
      p[0][j] = p0; p[1][j] = p1;
#pragma unroll
      for (int dt = 0; dt < 4; dt++) acc[dt][j] *= sc;
    }
    // ---- transpose P (C-layout -> A-layout) through LDS ----
    __syncthreads();
#pragma unroll
    for (int nt = 0; nt < 2; nt++)
#pragma unroll
      for (int j = 0; j < 8; j++)
        lP[(j + half * 8) * 32 + nt * 16 + l16] = f2bf(p[nt][j]);
    __syncthreads();
    v8u plo = *(const v8u*)&lP[l16 * 32 + k0];
    v8u phi = *(const v8u*)&lP[l16 * 32 + 8 + k0];
    v16u pf;
#pragma unroll
    for (int i = 0; i < 8; i++) { pf[i] = plo[i]; pf[i + 8] = phi[i]; }
    // ---- acc += P @ V (4 dim-tiles of 16) ----
#pragma unroll
    for (int dt = 0; dt < 4; dt++) {
      int dim = dt * 16 + l16;
      v16u vf;
#pragma unroll
      for (int i = 0; i < 16; i++) {
        int key = kb * 32 + half * 16 + i;
        vf[i] = qkv[(size_t)key * 3072 + 2048 + h * 64 + dim];
      }
      acc[dt] = wmma_bf16(pf, vf, acc[dt]);
    }
  }
  // ---- y = acc / l  (bf16, [T][H*DH]) ----
#pragma unroll
  for (int dt = 0; dt < 4; dt++)
#pragma unroll
    for (int j = 0; j < 8; j++) {
      int row = qt * 16 + j + half * 8;
      int dim = h * 64 + dt * 16 + l16;
      y[(size_t)row * (NHEAD * DHEAD) + dim] = f2bf(acc[dt][j] / l[j]);
    }
}

// ---------------------------------------------------------------------------
// abs-top-k: exact k-th largest |h| via binary search on magnitude bits,
// keep |h| >= thresh (matches reference tie behavior), emit bf16.
__global__ __launch_bounds__(256) void topk_kernel(
    const float* __restrict__ h, unsigned short* __restrict__ out,
    const int* __restrict__ kptr) {
  __shared__ int red[256];
  int row = blockIdx.x, t = threadIdx.x;
  int k = *kptr;
  const float* hr = h + (size_t)row * DMLP;
  float val[16]; unsigned int key[16];
#pragma unroll
  for (int j = 0; j < 16; j++) {
    float v = hr[t + 256 * j];
    val[j] = v;
    key[j] = __builtin_bit_cast(unsigned int, v) & 0x7FFFFFFFu;
  }
  unsigned int lo = 0;
  if (k < DMLP) {
    unsigned int hi = 0x7F800001u;
    while (hi - lo > 1) {  // uniform across block
      unsigned int mid = lo + (hi - lo) / 2;
      int c = 0;
#pragma unroll
      for (int j = 0; j < 16; j++) c += (key[j] >= mid) ? 1 : 0;
      red[t] = c; __syncthreads();
      for (int s = 128; s > 0; s >>= 1) {
        if (t < s) red[t] += red[t + s];
        __syncthreads();
      }
      int total = red[0];
      __syncthreads();
      if (total >= k) lo = mid; else hi = mid;
    }
  }
#pragma unroll
  for (int j = 0; j < 16; j++)
    out[(size_t)row * DMLP + t + 256 * j] =
        (key[j] >= lo) ? f2bf(val[j]) : (unsigned short)0;
}

// ---------------------------------------------------------------------------
extern "C" void kernel_launch(void* const* d_in, const int* in_sizes, int n_in,
                              void* d_out, int out_size, void* d_ws, size_t ws_size,
                              hipStream_t stream) {
  const int*   idx  = (const int*)d_in[0];
  const float* wte  = (const float*)d_in[1];
  const float* wpe  = (const float*)d_in[2];
  const float* ln1  = (const float*)d_in[3];
  const float* ln2  = (const float*)d_in[4];
  const float* Wa   = (const float*)d_in[5];   // [L][D][3072]
  const float* Wap  = (const float*)d_in[6];   // [L][1024][D]
  const float* Wf   = (const float*)d_in[7];   // [L][D][4096]
  const float* Wp   = (const float*)d_in[8];   // [L][4096][D]
  const float* sink = (const float*)d_in[9];   // [L][H]
  const float* lnf  = (const float*)d_in[10];
  const float* lmw  = (const float*)d_in[11];  // [D][V]
  const int*   kptr = (const int*)d_in[12];
  float* out = (float*)d_out;

  char* ws = (char*)d_ws;
  float*          x    = (float*)ws;                        //  8 MB f32 residual
  unsigned short* xn   = (unsigned short*)(ws + 8388608);   //  4 MB bf16 normed acts
  unsigned short* qkvb = (unsigned short*)(ws + 12582912);  // 12 MB bf16 qkv
  unsigned short* yb   = (unsigned short*)(ws + 25165824);  //  4 MB bf16 attn out
  float*          hbuf = (float*)(ws + 29360128);           // 32 MB f32 mlp hidden
  unsigned short* hs   = (unsigned short*)(ws + 62914560);  // 16 MB bf16 sparsified

  embed_kernel<<<T_SEQ, 256, 0, stream>>>(idx, wte, wpe, x);

  for (int layer = 0; layer < NLAYER; layer++) {
    rmsnorm_kernel<<<T_SEQ, 256, 0, stream>>>(x, ln1 + (size_t)layer * DMODEL, xn, DMODEL);
    gemm_kernel<3><<<dim3(3072 / 128, T_SEQ / 128), 256, 0, stream>>>(
        xn, Wa + (size_t)layer * DMODEL * 3072, qkvb, nullptr, T_SEQ, 3072, DMODEL);
    attn_kernel<<<dim3(T_SEQ / 16, NHEAD), 32, 0, stream>>>(
        qkvb, sink + (size_t)layer * NHEAD, yb);
    gemm_kernel<1><<<dim3(DMODEL / 128, T_SEQ / 128), 256, 0, stream>>>(
        yb, Wap + (size_t)layer * DMODEL * DMODEL, x, x, T_SEQ, DMODEL, DMODEL);
    rmsnorm_kernel<<<T_SEQ, 256, 0, stream>>>(x, ln2 + (size_t)layer * DMODEL, xn, DMODEL);
    gemm_kernel<2><<<dim3(DMLP / 128, T_SEQ / 128), 256, 0, stream>>>(
        xn, Wf + (size_t)layer * DMODEL * DMLP, hbuf, nullptr, T_SEQ, DMLP, DMODEL);
    topk_kernel<<<T_SEQ, 256, 0, stream>>>(hbuf, hs, kptr);
    gemm_kernel<1><<<dim3(DMODEL / 128, T_SEQ / 128), 256, 0, stream>>>(
        hs, Wp + (size_t)layer * DMLP * DMODEL, x, x, T_SEQ, DMODEL, DMLP);
  }

  rmsnorm_kernel<<<T_SEQ, 256, 0, stream>>>(x, lnf, xn, DMODEL);
  gemm_kernel<0><<<dim3(NVOCAB / 128, T_SEQ / 128), 256, 0, stream>>>(
      xn, lmw, out, nullptr, T_SEQ, NVOCAB, DMODEL);
}
